// VectorQuantizer_69458211111688
// MI455X (gfx1250) — matrix operand order repaired and checked by reading, not compiled, not run on hardware
//
#include <hip/hip_runtime.h>
#include <hip/hip_bf16.h>

// ---------------------------------------------------------------------------
// VQ-VAE vector quantizer for MI455X (gfx1250, wave32, WMMA + async-to-LDS).
//   z_e:        (32, 256, 32, 32) f32  -> N = 32768 rows of C = 256
//   emb_weight: (1024, 256) f32
// Score matmul (32768 x 1024 x 256) on v_wmma_f32_16x16x32_bf16.
// B (codebook) is double-buffered in LDS via GLOBAL_LOAD_ASYNC_TO_LDS_B128,
// and B fragments are double-buffered in registers so LDS latency overlaps
// WMMA issue (avoids s_wait_dscnt 0 before every WMMA pair).
// Each wave owns 32 rows (2 A row-sets in registers) -> every B fragment
// feeds 2 WMMAs; every staged B tile feeds all 4 waves (8x less L2 traffic).
// Gather + loss recomputed in exact fp32 from the original inputs.
// ---------------------------------------------------------------------------

typedef __attribute__((ext_vector_type(16))) __bf16 v16bf;
typedef __attribute__((ext_vector_type(8)))  __bf16 v8bf;
typedef __attribute__((ext_vector_type(8)))  float  v8f;

#define NUM_EMB   1024
#define EMB_DIM   256
#define HW        1024          // 32*32 spatial positions per image
#define ROWS_WG   128           // rows (spatial positions) per workgroup
#define ZQSTRIDE  72            // LDS z row stride, bf16 elems (64 + 8 pad)
#define EBSTRIDE  264           // LDS B col stride, bf16 elems (256 + 8 pad)
#define LOSS_OFF  8388608       // 32*256*32*32
#define IDX_OFF   8388609

// ---- CDNA5 async global->LDS copy (ASYNCcnt-tracked), builtin or asm ------
// Probe-confirmed signature: (int4 addrspace(1)*, int4 addrspace(3)*, imm, cpol)
typedef int v4i_vs __attribute__((__vector_size__(16)));
typedef __attribute__((address_space(1))) v4i_vs* gvec128_t;
typedef __attribute__((address_space(3))) v4i_vs* lvec128_t;

__device__ __forceinline__ void async_copy_b128(__bf16* lds_dst, const __bf16* gsrc) {
#if __has_builtin(__builtin_amdgcn_global_load_async_to_lds_b128)
    __builtin_amdgcn_global_load_async_to_lds_b128(
        (gvec128_t)(uintptr_t)gsrc,
        (lvec128_t)(unsigned)(uintptr_t)lds_dst, 0, 0);
#else
    unsigned l = (unsigned)(uintptr_t)lds_dst;   // LDS byte address = addr[31:0]
    asm volatile("global_load_async_to_lds_b128 %0, %1, off"
                 :: "v"(l), "v"(gsrc) : "memory");
#endif
}

__device__ __forceinline__ void wait_async0() {
#if __has_builtin(__builtin_amdgcn_s_wait_asynccnt)
    __builtin_amdgcn_s_wait_asynccnt(0);
#else
    asm volatile("s_wait_asynccnt 0x0" ::: "memory");
#endif
}

// ---------------- prep: codebook f32 -> bf16, and ||e||^2 per code ----------
__global__ __launch_bounds__(256) void vq_prep_emb(const float* __restrict__ emb,
                                                   __bf16* __restrict__ emb_bf,
                                                   float* __restrict__ e2) {
    const int code = blockIdx.x;      // 1024 blocks
    const int c    = threadIdx.x;     // 256 threads
    __shared__ float red[256];
    float w = emb[code * EMB_DIM + c];
    emb_bf[code * EMB_DIM + c] = (__bf16)w;
    red[c] = w * w;
    __syncthreads();
    for (int s = 128; s > 0; s >>= 1) {
        if (c < s) red[c] += red[c + s];
        __syncthreads();
    }
    if (c == 0) e2[code] = red[0];
}

// ---------------- zero the loss accumulator (per launch, replay-safe) -------
__global__ void vq_zero_loss(float* __restrict__ out) {
    out[LOSS_OFF] = 0.0f;
}

// ---------------- main kernel: 256 WGs x 128 threads (4 wave32) -------------
__global__ __launch_bounds__(128, 1)
void vq_main(const float* __restrict__ z,        // (32,256,32,32)
             const float* __restrict__ emb,      // (1024,256) f32
             const __bf16* __restrict__ emb_bf,  // (1024,256) bf16
             const float* __restrict__ e2,       // (1024) ||e||^2
             float* __restrict__ out) {
    __shared__ __bf16 Za[ROWS_WG * ZQSTRIDE];    // 18432 B (quarter-K z tile)
    __shared__ __bf16 Eb[2][32 * EBSTRIDE];      // 2 x 16896 B (B double buffer)
    __shared__ int    idxbuf[ROWS_WG];
    __shared__ float  lsum;

    const int tid  = threadIdx.x;
    const int wg   = blockIdx.x;                 // 256 WGs
    const int b    = wg >> 3;
    const int p0   = (wg & 7) * ROWS_WG;
    const size_t zbase = (size_t)b * EMB_DIM * HW + p0;

    if (tid == 0) lsum = 0.0f;

    const int lane    = tid & 31;
    const int wave    = tid >> 5;                // 0..3
    const int rowbase = wave * 32;               // 32 rows per wave
    const int halfk   = lane >> 4;               // 0: lanes 0-15, 1: lanes 16-31
    const int lcol    = lane & 15;

    // ---- kick off async staging of B group 0 (overlaps z staging) ----------
    // Group = 32 codes x 256 K bf16 (16 KB). Each thread copies 8 x 16B chunks
    // (chunk -> col_local = chunk/32, k = (chunk%32)*8): contiguous per thread.
    #pragma unroll
    for (int i = 0; i < 8; ++i) {
        const int chunk = tid * 8 + i;
        const int cl = chunk >> 5;
        const int k  = (chunk & 31) << 3;
        async_copy_b128(&Eb[0][cl * EBSTRIDE + k],
                        emb_bf + (size_t)cl * EMB_DIM + k);
    }

    // ---- stage z -> LDS bf16 in 4 K-quarters; A fragments -> registers -----
    // A k order used here (contiguous, same as B): lanes 0-15 row M=lane hold K 0..15;
    //                          lanes 16-31 row M=lane-16 hold K 16..31.
    v16bf afrag[2][8];                           // 2 row-sets x 8 K-steps
    for (int q = 0; q < 4; ++q) {
        if (q) __syncthreads();                  // protect previous quarter reads
        {
            const int r = tid;                   // 0..127
            #pragma unroll 8
            for (int cl = 0; cl < 64; ++cl) {
                float v = z[zbase + (size_t)(q * 64 + cl) * HW + r];
                Za[r * ZQSTRIDE + cl] = (__bf16)v;
            }
        }
        __syncthreads();
        #pragma unroll
        for (int rs = 0; rs < 2; ++rs) {
            const __bf16* ap = &Za[(rowbase + rs * 16 + lcol) * ZQSTRIDE + halfk * 16];
            #pragma unroll
            for (int kq = 0; kq < 2; ++kq) {
                v8bf lo = *(const v8bf*)(ap + kq * 32);
                v8bf hi = *(const v8bf*)(ap + kq * 32 + 8);
                afrag[rs][q * 2 + kq] = __builtin_shufflevector(
                    lo, hi, 0,1,2,3,4,5,6,7,8,9,10,11,12,13,14,15);
            }
        }
    }

    // ---- column sweep: 32 groups of 32 codes, double-buffered B in LDS -----
    float best[2][8];
    int   bidx[2][8];
    #pragma unroll
    for (int rs = 0; rs < 2; ++rs)
        #pragma unroll
        for (int v = 0; v < 8; ++v) { best[rs][v] = 3.4e38f; bidx[rs][v] = 0; }

    for (int g = 0; g < 32; ++g) {
        wait_async0();                           // buf[g&1] staged (this wave)
        __syncthreads();                         // visible to all; prev compute done
        if (g + 1 < 32) {                        // prefetch next group
            const int n0n = (g + 1) * 32;
            __bf16* dst = Eb[(g + 1) & 1];
            #pragma unroll
            for (int i = 0; i < 8; ++i) {
                const int chunk = tid * 8 + i;
                const int cl = chunk >> 5;
                const int k  = (chunk & 31) << 3;
                async_copy_b128(dst + cl * EBSTRIDE + k,
                                emb_bf + (size_t)(n0n + cl) * EMB_DIM + k);
            }
        }

        // B k order used here (contiguous, same as A): lanes 0-15 hold K 0..15 of col,
        // lanes 16-31 hold K 16..31; col = t*16 + (lane&15).
        const __bf16* ebbase = Eb[g & 1] + lcol * EBSTRIDE + halfk * 16;

        v8f acc[2][2];                           // [tile][rowset]
        #pragma unroll
        for (int t = 0; t < 2; ++t) { acc[t][0] = {}; acc[t][1] = {}; }

        // Software-pipelined B fragments: prefetch (ks,t)+1 into the alternate
        // register pair before the WMMAs consuming (ks,t). LDS returns in
        // order, so the consumer waits with dscnt<=2 instead of dscnt==0.
        v8bf plo = *(const v8bf*)(ebbase);
        v8bf phi = *(const v8bf*)(ebbase + 8);
        #pragma unroll
        for (int it = 0; it < 16; ++it) {        // it = ks*2 + t
            const int ks = it >> 1;
            const int t  = it & 1;
            v8bf lo = plo, hi = phi;
            if (it + 1 < 16) {
                const int nks = (it + 1) >> 1;
                const int nt  = (it + 1) & 1;
                const __bf16* np = ebbase + nt * (16 * EBSTRIDE) + nks * 32;
                plo = *(const v8bf*)(np);
                phi = *(const v8bf*)(np + 8);
            }
            v16bf bfrag = __builtin_shufflevector(
                lo, hi, 0,1,2,3,4,5,6,7,8,9,10,11,12,13,14,15);
            acc[t][0] = __builtin_amdgcn_wmma_f32_16x16x32_bf16(
                false, afrag[0][ks], false, bfrag, (short)0, acc[t][0], false, false);
            acc[t][1] = __builtin_amdgcn_wmma_f32_16x16x32_bf16(
                false, afrag[1][ks], false, bfrag, (short)0, acc[t][1], false, false);
        }

        // score = ||e||^2 - 2 z.e ; ascending cols + strict < => first argmin
        const int n0 = g * 32;
        #pragma unroll
        for (int t = 0; t < 2; ++t) {
            const int   col = n0 + t * 16 + lcol;
            const float ee  = e2[col];
            #pragma unroll
            for (int rs = 0; rs < 2; ++rs)
                #pragma unroll
                for (int v = 0; v < 8; ++v) {
                    float cand = ee - 2.0f * acc[t][rs][v];
                    if (cand < best[rs][v]) { best[rs][v] = cand; bidx[rs][v] = col; }
                }
        }
    }

    // ---- cross-lane argmin within each 16-lane half (xor butterfly) --------
    // D layout: lanes 0-15 hold rows (base + v), lanes 16-31 rows (base+8+v).
    #pragma unroll
    for (int rs = 0; rs < 2; ++rs)
        #pragma unroll
        for (int v = 0; v < 8; ++v)
            #pragma unroll
            for (int m = 1; m < 16; m <<= 1) {
                float os = __shfl_xor(best[rs][v], m, 32);
                int   oi = __shfl_xor(bidx[rs][v], m, 32);
                if (os < best[rs][v] || (os == best[rs][v] && oi < bidx[rs][v])) {
                    best[rs][v] = os; bidx[rs][v] = oi;
                }
            }
    if (lcol == 0) {
        #pragma unroll
        for (int rs = 0; rs < 2; ++rs)
            #pragma unroll
            for (int v = 0; v < 8; ++v)
                idxbuf[rowbase + rs * 16 + halfk * 8 + v] = bidx[rs][v];
    }
    __syncthreads();

    // ---- gather z_q (exact fp32), write output + indices, accumulate loss --
    float partial = 0.0f;
    {
        const int r = tid;                       // 0..127
        const int code = idxbuf[r];
        const float* erow = emb + (size_t)code * EMB_DIM;
        #pragma unroll 4
        for (int c = 0; c < EMB_DIM; ++c) {
            float e  = erow[c];
            float zv = z[zbase + (size_t)c * HW + r];
            float df = e - zv;
            partial += df * df;
            out[zbase + (size_t)c * HW + r] = e; // z_q in (B,C,H,W)
        }
        out[IDX_OFF + b * HW + p0 + r] = (float)code;
    }
    #pragma unroll
    for (int m = 16; m > 0; m >>= 1) partial += __shfl_xor(partial, m, 32);
    if (lane == 0) atomicAdd(&lsum, partial);
    __syncthreads();
    if (tid == 0) {
        // vq_loss = (1 + 0.25) * mean((z_q - z)^2), mean over 8388608 elems
        atomicAdd(out + LOSS_OFF, lsum * (1.25f / 8388608.0f));
    }
}

// ---------------------------------------------------------------------------
extern "C" void kernel_launch(void* const* d_in, const int* in_sizes, int n_in,
                              void* d_out, int out_size, void* d_ws, size_t ws_size,
                              hipStream_t stream) {
    const float* z_e = (const float*)d_in[0];    // 32*256*32*32
    const float* emb = (const float*)d_in[1];    // 1024*256
    float* out = (float*)d_out;

    __bf16* emb_bf = (__bf16*)d_ws;              // 512 KB
    float*  e2     = (float*)((char*)d_ws + (size_t)NUM_EMB * EMB_DIM * sizeof(__bf16));

    vq_prep_emb<<<NUM_EMB, 256, 0, stream>>>(emb, emb_bf, e2);
    vq_zero_loss<<<1, 1, 0, stream>>>(out);
    vq_main<<<256, 128, 0, stream>>>(z_e, emb, emb_bf, e2, out);
}
